// CrossNetworkV2_31215822308169
// MI455X (gfx1250) — matrix operand order repaired and checked by reading, not compiled, not run on hardware
//
#include <hip/hip_runtime.h>
#include <stdint.h>

// ---------------------------------------------------------------------------
// DCNv2 cross network, fused 3 layers, split-bf16 (3-product) WMMA on gfx1250
// x0, xl: [B, F, D]  W: [L, F, F]  bias: [L, F, D]
//   per layer: xl = x0 * (W @ xl + bias) + xl
// ---------------------------------------------------------------------------

#define NLAYERS 3
#define F       40
#define D       64
#define BATCH   16384

#define SPW      4                 // samples per workgroup
#define COLS     (SPW * D)         // 256 columns per workgroup
#define NTHREADS 512               // 16 waves (wave32)
#define NWAVES   16

#define KPAD     64                // K (=F) padded to 2 x 32 for bf16 WMMA
#define XK       72                // per-column stride in bf16 elems (64 + 8 pad)
#define XK_U4    9                 // uint4 per column (72*2/16)
#define X0STRIDE 260               // fp32 x0 tile row stride (256 + 4 pad)

// LDS partition (bytes), all 16B aligned
#define OFF_XH   0                         // 256*72*2  = 36864
#define OFF_XLO  36864                     // 36864
#define OFF_WPK  73728                     // 3*2*3*2*32*8*4 = 36864
#define OFF_X0F  110592                    // 40*260*4 = 41600
#define SMEM_BYTES 152192

typedef __attribute__((ext_vector_type(16))) __bf16 v16bf;
typedef __attribute__((ext_vector_type(8)))  float  v8f;

__device__ __forceinline__ unsigned short f2bf(float f) {
  union { float f; unsigned u; } v; v.f = f;
  unsigned r = v.u + 0x7FFFu + ((v.u >> 16) & 1u);   // round-to-nearest-even
  return (unsigned short)(r >> 16);
}
__device__ __forceinline__ float bf2f(unsigned short h) {
  union { float f; unsigned u; } v; v.u = ((unsigned)h) << 16;
  return v.f;
}

// Load a 16-element bf16 fragment (8 dwords) from LDS as two 16B loads.
__device__ __forceinline__ v16bf load_frag(const uint4* base, int idx4) {
  union { uint4 q[2]; v16bf v; } u;
  u.q[0] = base[idx4];
  u.q[1] = base[idx4 + 1];
  return u.v;
}

__global__ void __launch_bounds__(NTHREADS)
dcnv2_cross_fused(const float* __restrict__ x0,
                  const float* __restrict__ W,     // [3][40][40]
                  const float* __restrict__ bias,  // [3][40][64]
                  float* __restrict__ out)         // [B][40][64]
{
  extern __shared__ char smem[];
  unsigned short* xh   = (unsigned short*)(smem + OFF_XH);   // col-major bf16 hi
  unsigned short* xlo  = (unsigned short*)(smem + OFF_XLO);  // col-major bf16 lo
  unsigned*       wpk  = (unsigned*)      (smem + OFF_WPK);  // packed A frags
  float*          x0f  = (float*)         (smem + OFF_X0F);  // fp32 x0 tile

  const int tid  = threadIdx.x;
  const int wave = tid >> 5;
  const int lane = tid & 31;
  const int hi   = lane >> 4;          // lane half (0/1)
  const int ln16 = lane & 15;
  const int b0   = blockIdx.x * SPW;

  // ---- 1) zero xh/xlo (so padded K rows 40..63 contribute 0) --------------
  {
    uint4* p = (uint4*)smem;           // covers xh + xlo = 4608 uint4
    uint4 z = make_uint4(0u, 0u, 0u, 0u);
    for (int i = tid; i < (OFF_WPK / 16); i += NTHREADS) p[i] = z;
  }
  __syncthreads();

  // ---- 2) pack W (all layers, hi+lo splits) into A-fragment layout --------
  // idx = ((((layer*2+split)*3 + mt)*2 + kk)*32 + lane)*8 + v
// A 16x32 bf16 layout: M = lane&15 (+16*mt); lane<16: K in {0..15},
// lane>=16: K in {16..31}; VGPR v holds pair (K0, K0+1) packed.
  for (int i = tid; i < 9216; i += NTHREADS) {
    int v     = i & 7;
    int ln    = (i >> 3) & 31;
    int kk    = (i >> 8) & 1;
    int mt    = (i >> 9) % 3;
    int sl    = (i >> 9) / 3;          // layer*2 + split
    int split = sl & 1;
    int layer = sl >> 1;
    int lh    = ln >> 4;
    int M     = (ln & 15) + 16 * mt;
    int K0    = 2 * v + 16 * lh + kk * 32;
    float w0 = (M < F && (K0    ) < F) ? W[layer * F * F + M * F + K0    ] : 0.f;
    float w1 = (M < F && (K0 + 1) < F) ? W[layer * F * F + M * F + K0 + 1] : 0.f;
    unsigned short h0, h1;
    if (split == 0) { h0 = f2bf(w0); h1 = f2bf(w1); }
    else            { h0 = f2bf(w0 - bf2f(f2bf(w0)));
                      h1 = f2bf(w1 - bf2f(f2bf(w1))); }
    wpk[i] = (unsigned)h0 | ((unsigned)h1 << 16);
  }

  // ---- 3) stage x0 tile: fp32 row-major + bf16 hi/lo col-major ------------
  // 160 segments of 64 contiguous floats, 16 lanes x float4 per segment
  for (int seg = tid >> 4; seg < F * SPW; seg += (NTHREADS >> 4)) {
    int l16 = tid & 15;
    int s = seg / F;
    int g = seg % F;
    const float4 val =
        ((const float4*)(x0 + ((size_t)(b0 + s) * F + g) * D))[l16];
    int col = s * D + l16 * 4;
    ((float4*)(x0f + g * X0STRIDE + col))[0] = val;
    float vv[4] = {val.x, val.y, val.z, val.w};
#pragma unroll
    for (int q = 0; q < 4; ++q) {
      unsigned short h = f2bf(vv[q]);
      xh [(col + q) * XK + g] = h;
      xlo[(col + q) * XK + g] = f2bf(vv[q] - bf2f(h));
    }
  }
  __syncthreads();

  // ---- 4) fused layer loop -----------------------------------------------
  const uint4* xh4  = (const uint4*)xh;
  const uint4* xlo4 = (const uint4*)xlo;
  const uint4* wpk4 = (const uint4*)wpk;
  const int colbase = wave * 16;       // 16 N-columns per wave
  const int mycol   = colbase + ln16;

  for (int layer = 0; layer < NLAYERS; ++layer) {
    v8f acc[3];
#pragma unroll
    for (int mt = 0; mt < 3; ++mt) acc[mt] = (v8f)(0.f);

#pragma unroll
    for (int kk = 0; kk < 2; ++kk) {
      // B fragments: 16 consecutive-K bf16 at my column
      int bidx = mycol * XK_U4 + kk * 4 + hi * 2;
      v16bf Bh = load_frag(xh4,  bidx);
      v16bf Bl = load_frag(xlo4, bidx);
#pragma unroll
      for (int mt = 0; mt < 3; ++mt) {
        int ah = (((((layer * 2 + 0) * 3 + mt) * 2 + kk) * 32) + lane) * 2;
        int al = (((((layer * 2 + 1) * 3 + mt) * 2 + kk) * 32) + lane) * 2;
        v16bf Ah = load_frag(wpk4, ah);
        v16bf Al = load_frag(wpk4, al);
        // split-bf16: hi*hi + hi*lo + lo*hi, fp32 accumulate
        acc[mt] = __builtin_amdgcn_wmma_f32_16x16x32_bf16(
            false, Ah, false, Bh, (short)0, acc[mt], false, false);
        acc[mt] = __builtin_amdgcn_wmma_f32_16x16x32_bf16(
            false, Ah, false, Bl, (short)0, acc[mt], false, false);
        acc[mt] = __builtin_amdgcn_wmma_f32_16x16x32_bf16(
            false, Al, false, Bh, (short)0, acc[mt], false, false);
      }
    }
    __syncthreads();   // all WMMA reads of xh/xlo complete before update

    // elementwise: xl = x0 * (lw + bias) + xl  (C layout: vgpr j -> row j+8*hi)
    const int col  = mycol;
    const int s    = col >> 6;
    const int d    = col & 63;
    const bool last = (layer == NLAYERS - 1);
#pragma unroll
    for (int mt = 0; mt < 3; ++mt) {
#pragma unroll
      for (int j = 0; j < 8; ++j) {
        int f = 16 * mt + 8 * hi + j;
        if (f < F) {
          float lw  = acc[mt][j] + bias[layer * F * D + f * D + d];
          float xov = x0f[f * X0STRIDE + col];
          float xlv = bf2f(xh[col * XK + f]) + bf2f(xlo[col * XK + f]);
          float xn  = fmaf(xov, lw, xlv);
          if (last) {
            out[((size_t)(b0 + s) * F + f) * D + d] = xn;
          } else {
            unsigned short h = f2bf(xn);
            xh [col * XK + f] = h;
            xlo[col * XK + f] = f2bf(xn - bf2f(h));
          }
        }
      }
    }
    __syncthreads();   // updates visible before next layer's WMMA
  }
}

extern "C" void kernel_launch(void* const* d_in, const int* in_sizes, int n_in,
                              void* d_out, int out_size, void* d_ws, size_t ws_size,
                              hipStream_t stream) {
  const float* x0 = (const float*)d_in[0];
  const float* W  = (const float*)d_in[1];
  const float* bb = (const float*)d_in[2];
  float* out = (float*)d_out;

  // allow >64KB dynamic LDS (deterministic, host-side, capture-safe)
  hipFuncSetAttribute((const void*)dcnv2_cross_fused,
                      hipFuncAttributeMaxDynamicSharedMemorySize, SMEM_BYTES);

  dcnv2_cross_fused<<<BATCH / SPW, NTHREADS, SMEM_BYTES, stream>>>(x0, W, bb, out);
}
